// SelfAttentionModule_29042568855752
// MI455X (gfx1250) — matrix operand order repaired and checked by reading, not compiled, not run on hardware
//
#include <hip/hip_runtime.h>
#include <hip/hip_bf16.h>
#include <stdint.h>

// Cosine-similarity Gram matrix on MI455X (gfx1250, wave32).
//   out[b,m,n] = <x[b,:,m], x[b,:,n]> / max(||x[b,:,m]|| * ||x[b,:,n]||, 1e-8)
// B=8, C=512, N=56*56=3136.
// Roofline: 80.5 GFLOP vs ~366 MB HBM traffic (~16 us @ 23.3 TB/s) -> bf16 WMMA
// (fp32 accum) so matrix-core time matches the unavoidable store time.
// Pass 1 computes fp32 norms AND writes a bf16 pre-transposed copy XT[b][n][c]
// to workspace (25.7 MB, L2-resident). Pass 2 then tiles XT with *byte-copy*
// loads: GLOBAL_LOAD_ASYNC_TO_LDS_B128 (ASYNCcnt) -> zero packing VALU in the
// hot loop, half the global bytes, 4 copy instructions per wave per K-step.

#define BATCH 8
#define CDIM  512
#define NDIM  3136
#define TILE  128
#define KSTEP 32
#define NSTEP (CDIM / KSTEP)
#define LDSS  40              // LDS row stride in bf16 elements = 80 bytes (16B multiple)
#define EPS   1e-8f

typedef __attribute__((ext_vector_type(16))) __bf16 v16bf;
typedef __attribute__((ext_vector_type(8)))  float  v8f;
typedef __attribute__((ext_vector_type(4)))  int    v4i;
typedef __attribute__((address_space(1))) v4i* as1_v4i_ptr;
typedef __attribute__((address_space(3))) v4i* as3_v4i_ptr;

#if __has_builtin(__builtin_amdgcn_global_load_async_to_lds_b128)
#define USE_ASYNC 1
#else
#define USE_ASYNC 0
#endif

__device__ __forceinline__ uint32_t pack2_bf16(float a, float b) {
#if __has_builtin(__builtin_amdgcn_cvt_pk_bf16_f32)
  typedef __attribute__((ext_vector_type(2))) __bf16 v2bf;
  v2bf p = __builtin_amdgcn_cvt_pk_bf16_f32(a, b);
  return __builtin_bit_cast(uint32_t, p);
#else
  // round-to-nearest-even fp32 -> bf16, packed lo|hi (one-time prep pass only)
  uint32_t ua = __builtin_bit_cast(uint32_t, a);
  uint32_t ub = __builtin_bit_cast(uint32_t, b);
  ua = (ua + 0x7FFFu + ((ua >> 16) & 1u)) >> 16;
  ub = (ub + 0x7FFFu + ((ub >> 16) & 1u)) >> 16;
  return ua | (ub << 16);
#endif
}

__device__ __forceinline__ void wait_async_copies() {
#if USE_ASYNC
#if __has_builtin(__builtin_amdgcn_s_wait_asynccnt)
  __builtin_amdgcn_s_wait_asynccnt(0);
#else
  asm volatile("s_wait_asynccnt 0x0" ::: "memory");
#endif
#endif
}

// Pass 1: per-column fp32 L2 norms + bf16 transposed copy XT[b][n][c].
// Reads all of x (warms the 192MB L2); writes 25.7MB bf16.
__global__ __launch_bounds__(256)
void prep_kernel(const float* __restrict__ x, float* __restrict__ norms,
                 uint16_t* __restrict__ xt) {
  int idx = blockIdx.x * 256 + threadIdx.x;          // b*N + n
  if (idx >= BATCH * NDIM) return;
  int b = idx / NDIM, n = idx - b * NDIM;
  const float* p = x + (size_t)b * CDIM * NDIM + n;
  uint32_t* row = (uint32_t*)(xt + (size_t)idx * CDIM);
  float s = 0.f;
#pragma unroll 4
  for (int c = 0; c < CDIM; c += 2) {
    float v0 = p[(size_t)(c    ) * NDIM];
    float v1 = p[(size_t)(c + 1) * NDIM];
    s += v0 * v0 + v1 * v1;
    row[c >> 1] = pack2_bf16(v0, v1);
  }
  norms[idx] = sqrtf(s);
}

// Pass 2: tiled XT * XT^T (i.e. X^T X) with bf16 WMMA, / norm outer product.
__global__ __launch_bounds__(256)
void cosine_gram_wmma(const uint16_t* __restrict__ xt,
                      const float* __restrict__ norms,
                      float* __restrict__ out) {
  const int b  = blockIdx.z;
  const int m0 = blockIdx.x * TILE;                  // output row tile (q index)
  const int n0 = blockIdx.y * TILE;                  // output col tile (k index)
  const float* nb = norms + b * NDIM;

  // Double-buffered LDS tiles, stored [row (m or n)][K], bf16, 80B padded stride.
  __shared__ __align__(16) uint32_t sA[2][TILE * LDSS / 2];  // 2 x 10240 B
  __shared__ __align__(16) uint32_t sB[2][TILE * LDSS / 2];  // 2 x 10240 B

  const int tid = threadIdx.x;

  // ---- copy mapping: 256 threads = 2 matrices x 128 rows; 4 x 16B chunks each ----
  const int mat = tid >> 7;                          // 0 -> A tile (m0), 1 -> B tile (n0)
  const int row = tid & 127;
  int gr = (mat ? n0 : m0) + row;
  if (gr >= NDIM) gr = NDIM - 1;                     // duplicate edge row; stores guarded
  const char* grow = (const char*)(xt + ((size_t)b * NDIM + gr) * CDIM);

  // ---- compute mapping: 8 waves, each owns a 32(m) x 64(n) sub-tile ----
  const int wave = tid >> 5;
  const int lane = tid & 31;
  const int half = lane >> 4;                        // lane 0-15 vs 16-31
  const int l16  = lane & 15;
  const int wm   = (wave & 3) * 32;                  // 2 WMMA tiles in m
  const int wn   = (wave >> 2) * 64;                 // 4 WMMA tiles in n

  v8f acc[2][4];
#pragma unroll
  for (int i = 0; i < 2; ++i)
#pragma unroll
    for (int j = 0; j < 4; ++j) acc[i][j] = (v8f){0.f,0.f,0.f,0.f,0.f,0.f,0.f,0.f};

  // Copy one 128x32 bf16 slab per matrix into LDS buffer `buf` for K-offset k.
  // Pure byte copy: XT row-major [n][c] == LDS [row][K].
  auto copyTiles = [&](int buf, int k) {
    char* lrow = (char*)(mat ? sB[buf] : sA[buf]) + row * 80;
    const char* g = grow + k * 2;
#pragma unroll
    for (int ch = 0; ch < 4; ++ch) {
#if USE_ASYNC
      __builtin_amdgcn_global_load_async_to_lds_b128(
          (as1_v4i_ptr)(g + ch * 16),
          (as3_v4i_ptr)(lrow + ch * 16),
          0, 0);
#else
      *(uint4*)(lrow + ch * 16) = *(const uint4*)(g + ch * 16);
#endif
    }
  };

  auto computeStep = [&](int buf) {
    const char* const ac = (const char*)sA[buf];
    const char* const bc = (const char*)sB[buf];
    // A 16x32: lane m=l16; VGPRv <- K = half*16 + 2v,2v+1 -> 32 contiguous bytes
    v16bf fA[2];
#pragma unroll
    for (int i = 0; i < 2; ++i) {
      const int m = wm + i * 16 + l16;
      union { uint4 q[2]; v16bf v; } u;
      u.q[0] = *(const uint4*)(ac + m * 80 + half * 32);
      u.q[1] = *(const uint4*)(ac + m * 80 + half * 32 + 16);
      fA[i] = u.v;
    }
    // B 32x16: lane n=l16; VGPRv <- K = half*16 + 2v,2v+1 -> 32 contiguous bytes
    v16bf fB[4];
#pragma unroll
    for (int j = 0; j < 4; ++j) {
      const int n = wn + j * 16 + l16;
      union { uint4 q[2]; v16bf v; } u;
      u.q[0] = *(const uint4*)(bc + n * 80 + half * 32);
      u.q[1] = *(const uint4*)(bc + n * 80 + half * 32 + 16);
      fB[j] = u.v;
    }
#pragma unroll
    for (int i = 0; i < 2; ++i)
#pragma unroll
      for (int j = 0; j < 4; ++j)
        acc[i][j] = __builtin_amdgcn_wmma_f32_16x16x32_bf16(
            false, fA[i], false, fB[j], (short)0, acc[i][j], false, false);
  };

  // Prologue: fill buffer 0.
  copyTiles(0, 0);

  // Main loop: one barrier per K-step; async copies fill the other buffer
  // while the 8 WMMAs consume the current one.
  for (int t = 0; t < NSTEP; ++t) {
    const int cur = t & 1;
    wait_async_copies();                 // our copies into buf cur are complete
    __syncthreads();                     // all waves' copies into buf cur visible
    if (t + 1 < NSTEP) copyTiles(1 - cur, (t + 1) * KSTEP);
    computeStep(cur);
  }

  // ---- epilogue: divide by norm outer product, guarded stores ----
  const size_t outb = (size_t)b * NDIM * NDIM;
#pragma unroll
  for (int j = 0; j < 4; ++j) {
    const int gn = n0 + wn + j * 16 + l16;
    if (gn >= NDIM) continue;
    const float kn = nb[gn];
#pragma unroll
    for (int i = 0; i < 2; ++i) {
      const int gmb = m0 + wm + i * 16 + 8 * half;   // D layout: VGPR r -> m = r + 8*half
#pragma unroll
      for (int r = 0; r < 8; ++r) {
        const int gm = gmb + r;
        if (gm < NDIM) {
          const float qn = nb[gm];
          const float d = fmaxf(qn * kn, EPS);
          out[outb + (size_t)gm * NDIM + gn] = acc[i][j][r] / d;
        }
      }
    }
  }
}

extern "C" void kernel_launch(void* const* d_in, const int* in_sizes, int n_in,
                              void* d_out, int out_size, void* d_ws, size_t ws_size,
                              hipStream_t stream) {
  (void)in_sizes; (void)n_in; (void)out_size; (void)ws_size;
  const float* x = (const float*)d_in[0];
  float* out     = (float*)d_out;
  // Workspace layout: [0, B*N) fp32 norms; then bf16 XT[b][n][c] (25.7 MB).
  float*    norms = (float*)d_ws;
  uint16_t* xt    = (uint16_t*)((char*)d_ws + (size_t)BATCH * NDIM * sizeof(float));

  prep_kernel<<<(BATCH * NDIM + 255) / 256, 256, 0, stream>>>(x, norms, xt);

  const int tiles = (NDIM + TILE - 1) / TILE;        // 25
  dim3 grid(tiles, tiles, BATCH);
  cosine_gram_wmma<<<grid, 256, 0, stream>>>(xt, norms, out);
}
